// DotAlphaModule_89979564851543
// MI455X (gfx1250) — matrix operand order repaired and checked by reading, not compiled, not run on hardware
//
#include <hip/hip_runtime.h>
#include <hip/hip_bf16.h>

typedef __attribute__((ext_vector_type(16))) _Float16 v16h;
typedef __attribute__((ext_vector_type(8)))  _Float16 v8h;
typedef __attribute__((ext_vector_type(8)))  float    v8f;
typedef __attribute__((ext_vector_type(4)))  int      v4i;

#define WMMA_F16(a,b,c) __builtin_amdgcn_wmma_f32_16x16x32_f16(false,(a),false,(b),(short)0,(c),false,false)

// Optional CDNA5 async global->LDS path (ASYNCcnt), with safe fallback.
// Probe result (round 2): builtin exists, takes 4 args, param0 is `int4*`.
#if defined(__has_builtin)
# if __has_builtin(__builtin_amdgcn_global_load_async_to_lds_b128) && \
     __has_builtin(__builtin_amdgcn_s_wait_asynccnt)
#  define USE_ASYNC_LDS 1
# endif
#endif

#ifdef USE_ASYNC_LDS
__device__ __forceinline__ void async_copy_16B(const void* g, void* l) {
  __builtin_amdgcn_global_load_async_to_lds_b128(
      (v4i*)(unsigned long long)g, (v4i*)l, 0, 0);
}
#endif

// ---------------- problem constants ----------------
constexpr int NNODE = 8192;
constexpr int KEDGE = 32;
constexpr int FEAT  = 768;   // 6 blocks of 128
constexpr int FCOUT = 256;   // HEADS*HDIM
constexpr int EPW   = 64;    // edges per workgroup (2 nodes)

// ---------------- WMMA fragment helpers ----------------
// A-fragment (16x32 f16, M x K) loaded from LDS, row-major with given stride.
// Lane layout (this code's contiguous k order, not the ISA table order): lane L -> row = L%16, half = L/16;
//   elems 0..15 : k = k0 + half*16 + e
__device__ __forceinline__ v16h lds_a_frag(const _Float16* base, int stride, int row0, int k0) {
  int lane = threadIdx.x & 31;
  int hf = lane >> 4, m = lane & 15;
  const _Float16* p = base + (row0 + m) * stride + k0 + hf * 16;
  v8h lo = *(const v8h*)(p);
  v8h hi = *(const v8h*)(p + 8);
  return __builtin_shufflevector(lo, hi, 0,1,2,3,4,5,6,7,8,9,10,11,12,13,14,15);
}

// B-fragment (32x16 f16, K x N) from packed global buffer: one 32B load per lane.
__device__ __forceinline__ v16h b_frag(const _Float16* packed, int tile) {
  int lane = threadIdx.x & 31;
  return *(const v16h*)(packed + ((size_t)tile * 32 + lane) * 16);
}

// ---------------- weight packing ----------------
// Pack W (Cout x Cin, row-major fp32) into f16 B-fragments.
// tile = nt*KT + kt ; within tile: lane L holds n = nt*16 + L%16,
// elem e holds k = kt*32 + (L/16)*16 + e.
__device__ void pack_b(const float* W, int Cout, int Cin, _Float16* dst, int t, int S) {
  int KT = Cin >> 5;
  int total = (Cout >> 4) * KT * 512;
  for (int i = t; i < total; i += S) {
    int e = i & 15, lane = (i >> 4) & 31, tile = i >> 9;
    int kt = tile % KT, nt = tile / KT;
    int n = nt * 16 + (lane & 15);
    int k = kt * 32 + (lane >> 4) * 16 + e;
    dst[i] = (_Float16)W[(size_t)n * Cin + k];
  }
}
__device__ void pack_b_hilo(const float* W, int Cout, int Cin, _Float16* dh, _Float16* dl, int t, int S) {
  int KT = Cin >> 5;
  int total = (Cout >> 4) * KT * 512;
  for (int i = t; i < total; i += S) {
    int e = i & 15, lane = (i >> 4) & 31, tile = i >> 9;
    int kt = tile % KT, nt = tile / KT;
    int n = nt * 16 + (lane & 15);
    int k = kt * 32 + (lane >> 4) * 16 + e;
    float v = W[(size_t)n * Cin + k];
    _Float16 h = (_Float16)v;
    dh[i] = h;
    dl[i] = (_Float16)(v - (float)h);
  }
}

__global__ __launch_bounds__(256) void pack_kernel(
    const float* __restrict__ rw0, const float* __restrict__ rw1,
    const float* __restrict__ rw2, const float* __restrict__ fcw,
    const float* __restrict__ dotw,
    _Float16* p_r0, _Float16* p_r1, _Float16* p_r2, _Float16* p_fc,
    _Float16* p_dh, _Float16* p_dl) {
  int t = blockIdx.x * 256 + threadIdx.x;
  int S = gridDim.x * 256;
  pack_b(rw0,  64, 128, p_r0, t, S);
  pack_b(rw1,  64,  64, p_r1, t, S);
  pack_b(rw2, 768,  64, p_r2, t, S);
  pack_b(fcw, 256, 768, p_fc, t, S);
  for (int l = 0; l < 3; ++l)
    pack_b_hilo(dotw + (size_t)l * 16384, 128, 128,
                p_dh + (size_t)l * 16384, p_dl + (size_t)l * 16384, t, S);
}

// ---------------- node projection (split-precision f16 WMMA) ----------------
// proj[n, l*l+m, d] = sum_c node[n, l*l+m, c] * dot_w[l][d][c] (+ dot_b for l==0)
__global__ __launch_bounds__(256) void proj_kernel(
    const float* __restrict__ node, const float* __restrict__ dot_b,
    const _Float16* __restrict__ p_dh, const _Float16* __restrict__ p_dl,
    float* __restrict__ proj) {
  __shared__ _Float16 s_hi[16 * 128];
  __shared__ _Float16 s_lo[16 * 128];
  int bx = blockIdx.x, t = threadIdx.x;
  int l, rt;
  if (bx < 512)       { l = 0; rt = bx;        }
  else if (bx < 2048) { l = 1; rt = bx - 512;  }
  else                { l = 2; rt = bx - 2048; }
  const int M = 2 * l + 1, l2 = l * l;

  // stage 16 rows x 128, split hi/lo f16
  for (int i = t; i < 2048; i += 256) {
    int r = i >> 7, c = i & 127;
    int gr = rt * 16 + r;
    int n = gr / M, ml = gr - n * M;
    float v = node[((size_t)n * 9 + l2 + ml) * 128 + c];
    _Float16 h = (_Float16)v;
    s_hi[i] = h;
    s_lo[i] = (_Float16)(v - (float)h);
  }
  __syncthreads();

  int wv = t >> 5, lane = t & 31, ln = lane & 15, hf = lane >> 4;
  const _Float16* bh = p_dh + (size_t)l * 16384;
  const _Float16* bl = p_dl + (size_t)l * 16384;
  v8f acc = {};
  for (int kt = 0; kt < 4; ++kt) {
    v16h Ah = lds_a_frag(s_hi, 128, 0, kt * 32);
    v16h Al = lds_a_frag(s_lo, 128, 0, kt * 32);
    v16h Bh = b_frag(bh, wv * 4 + kt);
    v16h Bl = b_frag(bl, wv * 4 + kt);
    acc = WMMA_F16(Ah, Bh, acc);
    acc = WMMA_F16(Al, Bh, acc);
    acc = WMMA_F16(Ah, Bl, acc);
  }
  int col = wv * 16 + ln;
  float bias = (l == 0) ? dot_b[col] : 0.f;
#pragma unroll
  for (int r = 0; r < 8; ++r) {
    int gr = rt * 16 + r + 8 * hf;
    int n = gr / M, ml = gr - n * M;
    proj[((size_t)n * 9 + l2 + ml) * 128 + col] = acc[r] + bias;
  }
}

// ---------------- fused edge kernel ----------------
// LDS layout (bytes): s_x0 persistent, union region overlaid by lifetime.
constexpr int SM_X0   = 0;                 // f16 [64][768] = 98304 (x0, then x0*edge_m0)
constexpr int SM_U    = 98304;
constexpr int SM_PROJ = SM_U + 0;          // float [2][9][128] = 9216
constexpr int SM_SH   = SM_U + 9216;       // float [64][9]     = 2304
constexpr int SM_IDX  = SM_U + 11520;      // int   [64]        = 256
constexpr int SM_XE   = SM_U + 11776;      // f16   [64][128]   = 16384
constexpr int SM_H    = SM_U + 28160;      // float [64][64]    = 16384 (alias: fc B-stage f16)
constexpr int SM_HF   = SM_U + 44544;      // f16   [64][64]    = 8192
constexpr int SM_O    = SM_U + 0;          // f16   [64][256]   = 32768 (alias, after fc)
constexpr int SM_TOT  = SM_U + 52736;      // 151040 bytes total

__device__ __forceinline__ void ln_silu_row(const float* x, _Float16* o,
                                            const float* b, const float* g, const float* bb) {
  float mean = 0.f;
  for (int c = 0; c < 64; ++c) mean += x[c] + b[c];
  mean *= (1.f / 64.f);
  float var = 0.f;
  for (int c = 0; c < 64; ++c) { float d = x[c] + b[c] - mean; var += d * d; }
  var *= (1.f / 64.f);
  float rs = rsqrtf(var + 1e-5f);
  for (int c = 0; c < 64; ++c) {
    float v = (x[c] + b[c] - mean) * rs * g[c] + bb[c];
    float s = v / (1.f + __expf(-v));   // silu
    o[c] = (_Float16)s;
  }
}

__global__ __launch_bounds__(256) void edge_kernel(
    const float* __restrict__ xe, const float* __restrict__ ev,
    const int* __restrict__ gidx, const float* __restrict__ proj,
    const _Float16* __restrict__ p_r0, const _Float16* __restrict__ p_r1,
    const _Float16* __restrict__ p_r2, const _Float16* __restrict__ p_fc,
    const float* __restrict__ rb0, const float* __restrict__ rg0, const float* __restrict__ rbb0,
    const float* __restrict__ rb1, const float* __restrict__ rg1, const float* __restrict__ rbb1,
    const float* __restrict__ rb2, const float* __restrict__ fcb,
    const float* __restrict__ lng, const float* __restrict__ lnb,
    const float* __restrict__ adot, float* __restrict__ out) {
  __shared__ __align__(32) unsigned char smem[SM_TOT];
  _Float16* s_x0  = (_Float16*)(smem + SM_X0);
  float*    s_pj  = (float*)(smem + SM_PROJ);
  float*    s_sh  = (float*)(smem + SM_SH);
  int*      s_idx = (int*)(smem + SM_IDX);
  _Float16* s_xe  = (_Float16*)(smem + SM_XE);
  float*    s_h   = (float*)(smem + SM_H);
  _Float16* s_bst = (_Float16*)(smem + SM_H);   // alias (s_h dead during fc)
  _Float16* s_hf  = (_Float16*)(smem + SM_HF);
  _Float16* s_o   = (_Float16*)(smem + SM_O);   // alias (union dead after fc)

  const int t = threadIdx.x;
  const int wv = t >> 5, lane = t & 31, ln = lane & 15, hf = lane >> 4;
  const int n0 = blockIdx.x * 2;
  const size_t ebase = (size_t)n0 * KEDGE;

  // ---- stage proj(self), sh coeffs, gather indices, x_edge ----
  for (int i = t; i < 2 * 1152; i += 256) {
    int nn = i / 1152;
    s_pj[i] = proj[(size_t)(n0 + nn) * 1152 + (i - nn * 1152)];
  }
  if (t < EPW) {
    int k = t;
    const float* e3 = ev + (ebase + k) * 3;
    float x = e3[0], y = e3[1], z = e3[2];
    float r = fmaxf(sqrtf(x * x + y * y + z * z), 1e-12f);
    float inv = 1.f / r; x *= inv; y *= inv; z *= inv;
    float* sh = s_sh + k * 9;
    const float C1 = 0.4886025119029199f, C2 = 0.6307831305050401f, S3 = 1.7320508075688772f;
    sh[0] = 0.28209479177387814f;
    sh[1] = C1 * x; sh[2] = C1 * y; sh[3] = C1 * z;
    sh[4] = C2 * (S3 * x * z); sh[5] = C2 * (S3 * x * y);
    sh[6] = C2 * (y * y - 0.5f * (x * x + z * z));
    sh[7] = C2 * (S3 * y * z); sh[8] = C2 * (0.5f * S3 * (z * z - x * x));
    int gi = gidx[ebase + k];
    s_idx[k] = gi;
    const float* gp = proj + (size_t)gi * 1152;   // warm L2->L0 for the gather
#pragma unroll
    for (int m = 0; m < 9; ++m) __builtin_prefetch(gp + m * 128, 0, 3);
  }
  for (int i = t; i < EPW * 128; i += 256) s_xe[i] = (_Float16)xe[ebase * 128 + i];
  __syncthreads();

  // ---- x0: SH contraction (self + gathered), f16 into s_x0 ----
  for (int i = t; i < EPW * 128; i += 256) {
    int k = i >> 7, d = i & 127;
    const float* sp = s_pj + (k >> 5) * 1152;
    const float* gp = proj + (size_t)s_idx[k] * 1152;
    const float* sh = s_sh + k * 9;
    _Float16* xr = s_x0 + k * FEAT;
    xr[d]       = (_Float16)(sh[0] * sp[d]);
    xr[128 + d] = (_Float16)(sh[0] * gp[d]);
    float a1 = sh[1] * sp[128 + d] + sh[2] * sp[256 + d] + sh[3] * sp[384 + d];
    float g1 = sh[1] * gp[128 + d] + sh[2] * gp[256 + d] + sh[3] * gp[384 + d];
    xr[256 + d] = (_Float16)a1;
    xr[384 + d] = (_Float16)g1;
    float a2 = 0.f, g2 = 0.f;
#pragma unroll
    for (int m = 0; m < 5; ++m) {
      float c = sh[4 + m];
      a2 += c * sp[(4 + m) * 128 + d];
      g2 += c * gp[(4 + m) * 128 + d];
    }
    xr[512 + d] = (_Float16)a2;
    xr[640 + d] = (_Float16)g2;
  }
  __syncthreads();

  // ---- radial layer 0: [64x128] x [128->64] ----
  for (int tt = wv; tt < 16; tt += 8) {
    int rowt = tt & 3, colt = tt >> 2;
    v8f acc = {};
    for (int kt = 0; kt < 4; ++kt) {
      v16h a = lds_a_frag(s_xe, 128, rowt * 16, kt * 32);
      v16h b = b_frag(p_r0, colt * 4 + kt);
      acc = WMMA_F16(a, b, acc);
    }
#pragma unroll
    for (int r = 0; r < 8; ++r)
      s_h[(rowt * 16 + r + 8 * hf) * 64 + colt * 16 + ln] = acc[r];
  }
  __syncthreads();
  if (t < EPW) ln_silu_row(s_h + t * 64, s_hf + t * 64, rb0, rg0, rbb0);
  __syncthreads();

  // ---- radial layer 1: [64x64] x [64->64] ----
  for (int tt = wv; tt < 16; tt += 8) {
    int rowt = tt & 3, colt = tt >> 2;
    v8f acc = {};
    for (int kt = 0; kt < 2; ++kt) {
      v16h a = lds_a_frag(s_hf, 64, rowt * 16, kt * 32);
      v16h b = b_frag(p_r1, colt * 2 + kt);
      acc = WMMA_F16(a, b, acc);
    }
#pragma unroll
    for (int r = 0; r < 8; ++r)
      s_h[(rowt * 16 + r + 8 * hf) * 64 + colt * 16 + ln] = acc[r];
  }
  __syncthreads();
  if (t < EPW) ln_silu_row(s_h + t * 64, s_hf + t * 64, rb1, rg1, rbb1);
  __syncthreads();

  // ---- radial layer 2: [64x64] x [64->768]; multiply into x0 in-place ----
  for (int tt = wv; tt < 192; tt += 8) {
    int rowt = tt & 3, colt = tt >> 2;
    v8f acc = {};
    for (int kt = 0; kt < 2; ++kt) {
      v16h a = lds_a_frag(s_hf, 64, rowt * 16, kt * 32);
      v16h b = b_frag(p_r2, colt * 2 + kt);
      acc = WMMA_F16(a, b, acc);
    }
    int col = colt * 16 + ln;
    float bias = rb2[col];
#pragma unroll
    for (int r = 0; r < 8; ++r) {
      int row = rowt * 16 + r + 8 * hf;
      int o = row * FEAT + col;
      float m = acc[r] + bias;                 // edge_m0
      s_x0[o] = (_Float16)((float)s_x0[o] * m);
    }
  }
  __syncthreads();

  // ---- fc GEMM: [64x768] x [768->256], B staged in LDS per k-step ----
  {
    int rowt = wv & 3;
    int cbase = (wv >> 2) * 8;
    v8f acc[8] = {};
    for (int kt = 0; kt < 24; ++kt) {
      __syncthreads();                         // WAR on s_bst
#ifdef USE_ASYNC_LDS
      // 16 KB of B fragments, moved global->LDS without touching VGPRs.
      for (int c = t; c < 1024; c += 256) {    // 16B chunks
        int jj = c >> 6, wi = c & 63;
        async_copy_16B(p_fc + ((size_t)(jj * 24 + kt) * 512 + wi * 8),
                       s_bst + c * 8);
      }
      __builtin_amdgcn_s_wait_asynccnt(0);
#else
      for (int c = t; c < 512; c += 256) {     // 16 tiles x 32 lanes, 32B chunks
        int jj = c >> 5, L = c & 31;
        *(v16h*)(s_bst + c * 16) =
            *(const v16h*)(p_fc + ((size_t)(jj * 24 + kt) * 32 + L) * 16);
      }
#endif
      __syncthreads();
      v16h a = lds_a_frag(s_x0, FEAT, rowt * 16, kt * 32);
#pragma unroll
      for (int j = 0; j < 8; ++j) {
        v16h b = *(const v16h*)(s_bst + ((cbase + j) * 32 + lane) * 16);
        acc[j] = WMMA_F16(a, b, acc[j]);
      }
    }
    __syncthreads();                            // s_bst dead; s_o may overlap it
#pragma unroll
    for (int j = 0; j < 8; ++j) {
      int col = (cbase + j) * 16 + ln;
      float bias = fcb[col];
#pragma unroll
      for (int r = 0; r < 8; ++r) {
        int row = rowt * 16 + r + 8 * hf;
        s_o[row * FCOUT + col] = (_Float16)(acc[j][r] + bias);
      }
    }
  }
  __syncthreads();

  // ---- epilogue: LN(32) + smooth leaky relu + alpha dot ----
  for (int u = t; u < EPW * 8; u += 256) {
    int row = u >> 3, h = u & 7;
    const _Float16* x = s_o + row * FCOUT + h * 32;
    float mean = 0.f;
    for (int d = 0; d < 32; ++d) mean += (float)x[d];
    mean *= (1.f / 32.f);
    float var = 0.f;
    for (int d = 0; d < 32; ++d) { float dd = (float)x[d] - mean; var += dd * dd; }
    var *= (1.f / 32.f);
    float rs = rsqrtf(var + 1e-5f);
    float a = 0.f;
    for (int d = 0; d < 32; ++d) {
      float v = ((float)x[d] - mean) * rs * lng[d] + lnb[d];
      float sg = 1.f / (1.f + __expf(-v));
      float sl = 0.6f * v + 0.4f * v * (2.f * sg - 1.f);  // smooth leaky, slope 0.2
      a += sl * adot[h * 32 + d];
    }
    out[(ebase + row) * 8 + h] = a;
  }
}

// ---------------- host launcher ----------------
extern "C" void kernel_launch(void* const* d_in, const int* in_sizes, int n_in,
                              void* d_out, int out_size, void* d_ws, size_t ws_size,
                              hipStream_t stream) {
  const float* x_edge  = (const float*)d_in[0];
  const float* node    = (const float*)d_in[1];
  const float* edgev   = (const float*)d_in[2];
  const int*   idx     = (const int*)d_in[3];
  const float* dot_w   = (const float*)d_in[4];
  const float* dot_b   = (const float*)d_in[5];
  const float* rad_w0  = (const float*)d_in[6];
  const float* rad_b0  = (const float*)d_in[7];
  const float* rad_w1  = (const float*)d_in[8];
  const float* rad_b1  = (const float*)d_in[9];
  const float* rad_w2  = (const float*)d_in[10];
  const float* rad_b2  = (const float*)d_in[11];
  const float* rad_g0  = (const float*)d_in[12];
  const float* rad_bb0 = (const float*)d_in[13];
  const float* rad_g1  = (const float*)d_in[14];
  const float* rad_bb1 = (const float*)d_in[15];
  const float* fc_w    = (const float*)d_in[16];
  const float* fc_b    = (const float*)d_in[17];
  const float* ln_g    = (const float*)d_in[18];
  const float* ln_b    = (const float*)d_in[19];
  const float* adot    = (const float*)d_in[20];
  float* out = (float*)d_out;

  char* ws = (char*)d_ws;
  float* proj = (float*)ws;
  size_t off = (size_t)NNODE * 1152 * sizeof(float);       // 37,748,736
  _Float16* p_r0 = (_Float16*)(ws + off); off += (size_t)8192   * 2;
  _Float16* p_r1 = (_Float16*)(ws + off); off += (size_t)4096   * 2;
  _Float16* p_r2 = (_Float16*)(ws + off); off += (size_t)49152  * 2;
  _Float16* p_fc = (_Float16*)(ws + off); off += (size_t)196608 * 2;
  _Float16* p_dh = (_Float16*)(ws + off); off += (size_t)49152  * 2;
  _Float16* p_dl = (_Float16*)(ws + off); off += (size_t)49152  * 2;

  pack_kernel<<<256, 256, 0, stream>>>(rad_w0, rad_w1, rad_w2, fc_w, dot_w,
                                       p_r0, p_r1, p_r2, p_fc, p_dh, p_dl);
  proj_kernel<<<4608, 256, 0, stream>>>(node, dot_b, p_dh, p_dl, proj);
  edge_kernel<<<NNODE / 2, 256, 0, stream>>>(x_edge, edgev, idx, proj,
      p_r0, p_r1, p_r2, p_fc,
      rad_b0, rad_g0, rad_bb0, rad_b1, rad_g1, rad_bb1,
      rad_b2, fc_b, ln_g, ln_b, adot, out);
}
